// VisualDecoder_34428457845278
// MI455X (gfx1250) — matrix operand order repaired and checked by reading, not compiled, not run on hardware
//
#include <hip/hip_runtime.h>
#include <hip/hip_bf16.h>

// ---------------------------------------------------------------------------
// Types
// ---------------------------------------------------------------------------
typedef unsigned short u16;
typedef __attribute__((ext_vector_type(16))) __bf16 v16bf;
typedef __attribute__((ext_vector_type(8)))  float  v8f;
typedef __attribute__((ext_vector_type(4)))  unsigned int u32x4;

union Frag { v16bf v; u32x4 h[2]; u16 u[16]; };

__device__ __forceinline__ u16 f2bf(float x) {
    unsigned int u = __float_as_uint(x);
    u += 0x7FFFu + ((u >> 16) & 1u);     // round-to-nearest-even
    return (u16)(u >> 16);
}

__device__ __forceinline__ v8f vzero8() {
    v8f z;
#pragma unroll
    for (int i = 0; i < 8; ++i) z[i] = 0.0f;
    return z;
}

// ---------------------------------------------------------------------------
// fp32 -> bf16 conversions
// ---------------------------------------------------------------------------
__global__ void cvt_bf(const float* __restrict__ in, u16* __restrict__ out, long n) {
    long i = (long)blockIdx.x * blockDim.x + threadIdx.x;
    if (i < n) out[i] = f2bf(in[i]);
}

// qw/kw/vw: [L,H,D,E] -> [L, D, H*E]  (so projection is one [768x768] GEMM)
__global__ void cvt_qkvw(const float* __restrict__ in, u16* __restrict__ out, long n) {
    long i = (long)blockIdx.x * blockDim.x + threadIdx.x;
    if (i >= n) return;
    int  e = (int)(i % 64);
    long r = i / 64;
    int  d = (int)(r % 768); r /= 768;
    int  h = (int)(r % 12);
    int  l = (int)(r / 12);
    out[((long)(l * 768 + d)) * 768 + h * 64 + e] = f2bf(in[i]);
}

// queries = x + qc[l] ; emit fp32 + bf16
__global__ void add_qc(const float* __restrict__ x, const float* __restrict__ qc,
                       float* __restrict__ of, u16* __restrict__ ob,
                       long perB, long total) {
    long i = (long)blockIdx.x * blockDim.x + threadIdx.x;
    if (i >= total) return;
    float v = x[i] + qc[i % perB];
    of[i] = v;
    ob[i] = f2bf(v);
}

// ---------------------------------------------------------------------------
// Generic bf16 WMMA GEMM:  C[M,N] = A[M,K] * B[K,N] (+bias) (+gelu)
//   BM: block tile edge (BM x BM), 64 -> 4 waves, 128 -> 8 waves.
//   TRANSB: B stored [N,K] row-major.
//   Double-buffered LDS; A (and trans-B) staged with async-to-LDS loads
//   (ASYNCcnt), non-trans B register-staged with ds transpose scatter.
//   Batched via grid.z: z -> (zo = z/zmod, zi = z%zmod); offsets O*zo + I*zi
// ---------------------------------------------------------------------------
template <int EPI, bool TRANSB, int BM>
__global__ __launch_bounds__(BM * 2) void gemm_bf16(
    const u16* __restrict__ A, long aO, long aI,
    const u16* __restrict__ Bm, long bO, long bI,
    const float* __restrict__ bias, long biasO, long biasI,
    float* __restrict__ Cf, long cO, long cI,
    u16* __restrict__ Cb, long cbO, long cbI,
    int M, int N, int K, int lda, int ldb, int ldc, int zmod)
{
    constexpr int THREADS = BM * 2;
    constexpr int NT = (BM == 128) ? 4 : 2;    // 16-col tiles per wave
    constexpr int BN = BM;

    __shared__ __align__(16) u16 lA[2][BM][40];
    __shared__ __align__(16) u16 lB[2][BN][40];

    const int z = blockIdx.z, zo = z / zmod, zi = z % zmod;
    A  += zo * aO + zi * aI;
    Bm += zo * bO + zi * bI;
    if (bias) bias += zo * biasO + zi * biasI;
    if (Cf)   Cf   += zo * cO    + zi * cI;
    if (Cb)   Cb   += zo * cbO   + zi * cbI;

    const int nBase = blockIdx.x * BN;
    const int mBase = blockIdx.y * BM;
    const int tid  = threadIdx.x;
    const int wid  = tid >> 5;
    const int lane = tid & 31;
    const int lm   = lane & 15;
    const int hiq  = (lane >> 4) & 1;
    const int wM = (wid >> 1) * 32;            // 64: {0,32} ; 128: {0,32,64,96}
    const int wN = (wid & 1) * (16 * NT);      // 64: {0,32} ; 128: {0,64}

    // --- async stage of a [rows x 32] row-major tile into LDS buffer
    auto loadA = [&](int buf, int k0) {
#pragma unroll
        for (int it = 0; it < 2; ++it) {
            int s = tid + it * THREADS;
            int row = s >> 2, kc = (s & 3) * 8;
            unsigned lo = (unsigned)(unsigned long long)&lA[buf][row][kc];
            const u16* ga = A + (long)(mBase + row) * lda + k0 + kc;
            asm volatile("global_load_async_to_lds_b128 %0, %1, off"
                         :: "v"(lo), "v"(ga) : "memory");
        }
    };
    auto loadBt = [&](int buf, int k0) {       // TRANSB: B rows are out columns
#pragma unroll
        for (int it = 0; it < 2; ++it) {
            int s = tid + it * THREADS;
            int row = s >> 2, kc = (s & 3) * 8;
            unsigned lo = (unsigned)(unsigned long long)&lB[buf][row][kc];
            const u16* ga = Bm + (long)(nBase + row) * ldb + k0 + kc;
            asm volatile("global_load_async_to_lds_b128 %0, %1, off"
                         :: "v"(lo), "v"(ga) : "memory");
        }
    };
    // --- non-trans B: [32 x BN] global tile, register-staged, transposed store
    u32x4 breg[2];
    auto fetchB = [&](int k0) {
#pragma unroll
        for (int it = 0; it < 2; ++it) {
            int s = tid + it * THREADS;
            int kk = s / (BN / 8), nc = (s % (BN / 8)) * 8;
            breg[it] = *(const u32x4*)(Bm + (long)(k0 + kk) * ldb + nBase + nc);
        }
    };
    auto storeB = [&](int buf) {
#pragma unroll
        for (int it = 0; it < 2; ++it) {
            int s = tid + it * THREADS;
            int kk = s / (BN / 8), nc = (s % (BN / 8)) * 8;
            const u16* ts = (const u16*)&breg[it];
#pragma unroll
            for (int j = 0; j < 8; ++j) lB[buf][nc + j][kk] = ts[j];
        }
    };

    v8f acc[2][NT];
#pragma unroll
    for (int i = 0; i < 2; ++i)
#pragma unroll
        for (int j = 0; j < NT; ++j) acc[i][j] = vzero8();

    // --- prologue: stage K-tile 0 into buffer 0
    loadA(0, 0);
    if (TRANSB) loadBt(0, 0);
    else { fetchB(0); storeB(0); }
    asm volatile("s_wait_asynccnt 0x0" ::: "memory");
    __syncthreads();

    const int KT = K / 32;
    for (int kt = 0; kt < KT; ++kt) {
        const int cur = kt & 1;
        const bool more = (kt + 1 < KT);
        if (more) {                              // prefetch next tile
            loadA(1 - cur, (kt + 1) * 32);
            if (TRANSB) loadBt(1 - cur, (kt + 1) * 32);
            else fetchB((kt + 1) * 32);
        }

        Frag aF[2], bF[NT];
#pragma unroll
        for (int mt = 0; mt < 2; ++mt) {
            const u32x4* pa = (const u32x4*)(&lA[cur][wM + mt * 16 + lm][0]);
            aF[mt].h[0] = pa[hiq];               // K {0..7} or {8..15}
            aF[mt].h[1] = pa[hiq + 2];           // K {16..23} or {24..31}
        }
#pragma unroll
        for (int nt = 0; nt < NT; ++nt) {
            const u32x4* pb = (const u32x4*)(&lB[cur][wN + nt * 16 + lm][0]);
            bF[nt].h[0] = pb[hiq];               // K {0..7} or {8..15}
            bF[nt].h[1] = pb[hiq + 2];           // K {16..23} or {24..31}
        }
#pragma unroll
        for (int mt = 0; mt < 2; ++mt)
#pragma unroll
            for (int nt = 0; nt < NT; ++nt)
                acc[mt][nt] = __builtin_amdgcn_wmma_f32_16x16x32_bf16(
                    false, aF[mt].v, false, bF[nt].v,
                    (short)0, acc[mt][nt], false, false);

        if (more) {
            if (!TRANSB) storeB(1 - cur);        // scatter into other buffer
            asm volatile("s_wait_asynccnt 0x0" ::: "memory");
        }
        __syncthreads();
    }

    // --- epilogue
#pragma unroll
    for (int mt = 0; mt < 2; ++mt)
#pragma unroll
        for (int nt = 0; nt < NT; ++nt)
#pragma unroll
            for (int r = 0; r < 8; ++r) {
                int rg = mBase + wM + mt * 16 + r + hiq * 8;
                int cg = nBase + wN + nt * 16 + lm;
                float c = acc[mt][nt][r];
                if (bias) c += bias[cg];
                if (EPI == 1) c = 0.5f * c * (1.0f + erff(c * 0.70710678118654752f));
                long idx = (long)rg * ldc + cg;
                if (Cf) Cf[idx] = c;
                if (Cb) Cb[idx] = f2bf(c);
            }
}

// ---------------------------------------------------------------------------
// Flash attention: one WG per (b,h); 4 waves x 16 query rows; 64-key chunks.
// q: [B,64,768] bf16 (col = h*64+e); k,v: [B,4096,768] bf16; o: [B,H,64,64] bf16
// ---------------------------------------------------------------------------
__global__ __launch_bounds__(128) void flash_attn(
    const u16* __restrict__ Q, const u16* __restrict__ Kt,
    const u16* __restrict__ Vt, u16* __restrict__ O,
    int H, int N, float scale)
{
    __shared__ __align__(16) u16 lK[64][72];      // K chunk, natural [key][e]
    __shared__ __align__(16) u16 lV[64][72];      // V chunk, transposed [e][key]
    __shared__ __align__(16) u16 lP[4][16][72];   // per-wave P staging

    const int bh = blockIdx.x;
    const int b = bh / H, h = bh % H;
    const int tid  = threadIdx.x;
    const int wid  = tid >> 5;
    const int lane = tid & 31;
    const int lm   = lane & 15;
    const int hiq  = (lane >> 4) & 1;
    const int m0   = wid * 16;

    // preload q A-fragments (K-dim = 64 -> two 32-wide steps)
    Frag aq[2];
#pragma unroll
    for (int s = 0; s < 2; ++s) {
        const u32x4* pq = (const u32x4*)(Q + ((long)(b * 64 + m0 + lm)) * 768
                                           + h * 64 + s * 32 + hiq * 8);
        aq[s].h[0] = pq[0];
        aq[s].h[1] = pq[2];
    }

    v8f oacc[4];
#pragma unroll
    for (int i = 0; i < 4; ++i) oacc[i] = vzero8();
    float mrun[8], lrun[8];
#pragma unroll
    for (int r = 0; r < 8; ++r) { mrun[r] = -1e30f; lrun[r] = 0.0f; }

    for (int n0 = 0; n0 < N; n0 += 64) {
        // ---- stage K (natural) and V (transposed) chunks
#pragma unroll
        for (int it = 0; it < 4; ++it) {
            int s = tid + it * 128;
            int row = s >> 3, ec = (s & 7) * 8;
            long src = ((long)(b * N + n0 + row)) * 768 + h * 64 + ec;
            *(u32x4*)&lK[row][ec] = *(const u32x4*)(Kt + src);
            u32x4 vv = *(const u32x4*)(Vt + src);
            const u16* vs = (const u16*)&vv;
#pragma unroll
            for (int j = 0; j < 8; ++j) lV[ec + j][row] = vs[j];
        }
        __syncthreads();

        // ---- S = q * K^T  (16 x 64 per wave)
        v8f sacc[4];
#pragma unroll
        for (int i = 0; i < 4; ++i) sacc[i] = vzero8();
#pragma unroll
        for (int ks = 0; ks < 2; ++ks) {
#pragma unroll
            for (int nt = 0; nt < 4; ++nt) {
                Frag bk;
                const u32x4* pb = (const u32x4*)(&lK[nt * 16 + lm][0]);
                int o0 = ks * 4 + hiq;
                bk.h[0] = pb[o0];
                bk.h[1] = pb[o0 + 2];
                sacc[nt] = __builtin_amdgcn_wmma_f32_16x16x32_bf16(
                    false, aq[ks].v, false, bk.v, (short)0, sacc[nt], false, false);
            }
        }

        // ---- online softmax (row spread over 16 lanes of a half-wave)
        float mnew[8], psum[8];
#pragma unroll
        for (int r = 0; r < 8; ++r) {
            float mx = fmaxf(fmaxf(sacc[0][r], sacc[1][r]),
                             fmaxf(sacc[2][r], sacc[3][r]));
#pragma unroll
            for (int off = 1; off < 16; off <<= 1)
                mx = fmaxf(mx, __shfl_xor(mx, off, 32));
            mx *= scale;
            mnew[r] = fmaxf(mrun[r], mx);
            float alpha = __expf(mrun[r] - mnew[r]);
#pragma unroll
            for (int nt = 0; nt < 4; ++nt) oacc[nt][r] *= alpha;
            lrun[r] *= alpha;
            mrun[r] = mnew[r];
            psum[r] = 0.0f;
        }
#pragma unroll
        for (int nt = 0; nt < 4; ++nt)
#pragma unroll
            for (int r = 0; r < 8; ++r) {
                float p = __expf(sacc[nt][r] * scale - mnew[r]);
                psum[r] += p;
                lP[wid][r + hiq * 8][nt * 16 + lm] = f2bf(p);
            }
#pragma unroll
        for (int r = 0; r < 8; ++r) {
            float ps = psum[r];
#pragma unroll
            for (int off = 1; off < 16; off <<= 1)
                ps += __shfl_xor(ps, off, 32);
            lrun[r] += ps;
        }
        asm volatile("s_wait_dscnt 0x0" ::: "memory");  // wave-local P stores -> reads

        // ---- O += P * V
#pragma unroll
        for (int ks = 0; ks < 2; ++ks) {
            Frag ap;
            const u32x4* pp = (const u32x4*)(&lP[wid][lm][0]);
            int oa = ks * 4 + hiq;
            ap.h[0] = pp[oa];
            ap.h[1] = pp[oa + 2];
#pragma unroll
            for (int nt = 0; nt < 4; ++nt) {
                Frag bv;
                const u32x4* pb = (const u32x4*)(&lV[nt * 16 + lm][0]);
                int o0 = ks * 4 + hiq;
                bv.h[0] = pb[o0];
                bv.h[1] = pb[o0 + 2];
                oacc[nt] = __builtin_amdgcn_wmma_f32_16x16x32_bf16(
                    false, ap.v, false, bv.v, (short)0, oacc[nt], false, false);
            }
        }
        __syncthreads();
    }

    // ---- normalize + store o[b,h,m,e] bf16
#pragma unroll
    for (int nt = 0; nt < 4; ++nt)
#pragma unroll
        for (int r = 0; r < 8; ++r) {
            int row = m0 + r + hiq * 8;
            int col = nt * 16 + lm;
            float val = oacc[nt][r] / lrun[r];
            O[((long)bh * 64 + row) * 64 + col] = f2bf(val);
        }
}

// ---------------------------------------------------------------------------
// Row softmax for pooling weights: rows = B*M, N = 4096; out bf16
// ---------------------------------------------------------------------------
__global__ __launch_bounds__(256) void row_softmax(
    const float* __restrict__ in, u16* __restrict__ ob)
{
    __shared__ float red[8];
    const long row = blockIdx.x;
    const float* x = in + row * 4096;
    int t = threadIdx.x;
    float vals[16];
    float mx = -1e30f;
#pragma unroll
    for (int i = 0; i < 16; ++i) { vals[i] = x[t + i * 256]; mx = fmaxf(mx, vals[i]); }
#pragma unroll
    for (int off = 1; off < 32; off <<= 1) mx = fmaxf(mx, __shfl_xor(mx, off, 32));
    if ((t & 31) == 0) red[t >> 5] = mx;
    __syncthreads();
    float bm = red[0];
#pragma unroll
    for (int i = 1; i < 8; ++i) bm = fmaxf(bm, red[i]);
    float s = 0.0f;
#pragma unroll
    for (int i = 0; i < 16; ++i) { vals[i] = __expf(vals[i] - bm); s += vals[i]; }
#pragma unroll
    for (int off = 1; off < 32; off <<= 1) s += __shfl_xor(s, off, 32);
    __syncthreads();
    if ((t & 31) == 0) red[t >> 5] = s;
    __syncthreads();
    float bs = 0.0f;
#pragma unroll
    for (int i = 0; i < 8; ++i) bs += red[i];
    float inv = 1.0f / bs;
    u16* o = ob + row * 4096;
#pragma unroll
    for (int i = 0; i < 16; ++i) o[t + i * 256] = f2bf(vals[i] * inv);
}

// ---------------------------------------------------------------------------
// Fused residual add + LayerNorm; D = 768, 256 threads, 3 elems/thread
// ---------------------------------------------------------------------------
__global__ __launch_bounds__(256) void ln_res(
    const float* __restrict__ a, const float* __restrict__ res,
    const float* __restrict__ g, const float* __restrict__ be,
    float* __restrict__ of, u16* __restrict__ obf)
{
    __shared__ float red[8];
    const long row = blockIdx.x;
    int t = threadIdx.x;
    const float* pa = a + row * 768;
    const float* pr = res + row * 768;
    float v[3];
    float s = 0.0f;
#pragma unroll
    for (int i = 0; i < 3; ++i) { v[i] = pa[t + i * 256] + pr[t + i * 256]; s += v[i]; }
#pragma unroll
    for (int off = 1; off < 32; off <<= 1) s += __shfl_xor(s, off, 32);
    if ((t & 31) == 0) red[t >> 5] = s;
    __syncthreads();
    float mean = 0.0f;
#pragma unroll
    for (int i = 0; i < 8; ++i) mean += red[i];
    mean *= (1.0f / 768.0f);
    float vs = 0.0f;
#pragma unroll
    for (int i = 0; i < 3; ++i) { float d = v[i] - mean; vs += d * d; }
#pragma unroll
    for (int off = 1; off < 32; off <<= 1) vs += __shfl_xor(vs, off, 32);
    __syncthreads();
    if ((t & 31) == 0) red[t >> 5] = vs;
    __syncthreads();
    float var = 0.0f;
#pragma unroll
    for (int i = 0; i < 8; ++i) var += red[i];
    var *= (1.0f / 768.0f);
    float rinv = rsqrtf(var + 1e-5f);
#pragma unroll
    for (int i = 0; i < 3; ++i) {
        int c = t + i * 256;
        float y = (v[i] - mean) * rinv * g[c] + be[c];
        of[row * 768 + c] = y;
        if (obf) obf[row * 768 + c] = f2bf(y);
    }
}

// ---------------------------------------------------------------------------
// Host
// ---------------------------------------------------------------------------
extern "C" void kernel_launch(void* const* d_in, const int* in_sizes, int n_in,
                              void* d_out, int out_size, void* d_ws, size_t ws_size,
                              hipStream_t stream)
{
    (void)in_sizes; (void)n_in; (void)out_size; (void)ws_size;
    const int Bc = 8, Nc = 4096, Dc = 768, Hc = 12, Mc = 64, Lc = 4;

    const float* latent   = (const float*)d_in[0];
    const float* selector = (const float*)d_in[1];
    const float* qc  = (const float*)d_in[2];
    const float* qw  = (const float*)d_in[3];
    const float* qb  = (const float*)d_in[4];
    const float* kw  = (const float*)d_in[5];
    const float* kb  = (const float*)d_in[6];
    const float* vw  = (const float*)d_in[7];
    const float* vb  = (const float*)d_in[8];
    const float* hw  = (const float*)d_in[9];
    const float* hb  = (const float*)d_in[10];
    const float* ow  = (const float*)d_in[11];
    const float* obi = (const float*)d_in[12];
    const float* w1  = (const float*)d_in[13];
    const float* b1  = (const float*)d_in[14];
    const float* w2  = (const float*)d_in[15];
    const float* b2  = (const float*)d_in[16];
    const float* g1  = (const float*)d_in[17];
    const float* be1 = (const float*)d_in[18];
    const float* g2  = (const float*)d_in[19];
    const float* be2 = (const float*)d_in[20];

    // --- workspace carve-up (256B aligned)
    size_t off = 0;
    auto alloc = [&](size_t bytes) -> char* {
        char* p = (char*)d_ws + off;
        off = (off + bytes + 255) & ~(size_t)255;
        return p;
    };
    u16*   lat_b  = (u16*)  alloc((size_t)Bc * Nc * Dc * 2);
    u16*   sel_b  = (u16*)  alloc((size_t)Mc * Dc * 2);
    u16*   wq_b   = (u16*)  alloc((size_t)Lc * Dc * Dc * 2);
    u16*   wk_b   = (u16*)  alloc((size_t)Lc * Dc * Dc * 2);
    u16*   wv_b   = (u16*)  alloc((size_t)Lc * Dc * Dc * 2);
    u16*   whw_b  = (u16*)  alloc((size_t)Lc * Hc * 64 * 64 * 2);
    u16*   wow_b  = (u16*)  alloc((size_t)Lc * Dc * Dc * 2);
    u16*   w1_b   = (u16*)  alloc((size_t)Lc * Dc * 4 * Dc * 2);
    u16*   w2_b   = (u16*)  alloc((size_t)Lc * 4 * Dc * Dc * 2);
    float* logit_f= (float*)alloc((size_t)Bc * Mc * Nc * 4);
    u16*   poolw_b= (u16*)  alloc((size_t)Bc * Mc * Nc * 2);
    float* x_f    = (float*)alloc((size_t)Bc * Mc * Dc * 4);
    u16*   x_b    = (u16*)  alloc((size_t)Bc * Mc * Dc * 2);
    float* qry_f  = (float*)alloc((size_t)Bc * Mc * Dc * 4);
    u16*   qry_b  = (u16*)  alloc((size_t)Bc * Mc * Dc * 2);
    u16*   q_b    = (u16*)  alloc((size_t)Bc * Mc * Dc * 2);
    u16*   k_b    = (u16*)  alloc((size_t)Bc * Nc * Dc * 2);
    u16*   v_b    = (u16*)  alloc((size_t)Bc * Nc * Dc * 2);
    u16*   o_b    = (u16*)  alloc((size_t)Bc * Hc * Mc * 64 * 2);
    float* om_f   = (float*)alloc((size_t)Bc * Mc * Dc * 4);
    u16*   om_b   = (u16*)  alloc((size_t)Bc * Mc * Dc * 2);
    float* ao_f   = (float*)alloc((size_t)Bc * Mc * Dc * 4);
    float* x1_f   = (float*)alloc((size_t)Bc * Mc * Dc * 4);
    u16*   x1_b   = (u16*)  alloc((size_t)Bc * Mc * Dc * 2);
    u16*   mh_b   = (u16*)  alloc((size_t)Bc * Mc * 4 * Dc * 2);
    float* ml_f   = (float*)alloc((size_t)Bc * Mc * Dc * 4);

    auto cvtN = [&](const float* in, u16* out, long n) {
        cvt_bf<<<dim3((unsigned)((n + 255) / 256)), dim3(256), 0, stream>>>(in, out, n);
    };

    // --- one-time conversions (deterministic every call)
    cvtN(latent,   lat_b,  (long)Bc * Nc * Dc);
    cvtN(selector, sel_b,  (long)Mc * Dc);
    {
        long nq = (long)Lc * Hc * Dc * 64;
        cvt_qkvw<<<dim3((unsigned)((nq + 255) / 256)), dim3(256), 0, stream>>>(qw, wq_b, nq);
        cvt_qkvw<<<dim3((unsigned)((nq + 255) / 256)), dim3(256), 0, stream>>>(kw, wk_b, nq);
        cvt_qkvw<<<dim3((unsigned)((nq + 255) / 256)), dim3(256), 0, stream>>>(vw, wv_b, nq);
    }
    cvtN(hw, whw_b, (long)Lc * Hc * 64 * 64);
    cvtN(ow, wow_b, (long)Lc * Dc * Dc);
    cvtN(w1, w1_b,  (long)Lc * Dc * 4 * Dc);
    cvtN(w2, w2_b,  (long)Lc * 4 * Dc * Dc);

    // --- GEMM dispatch helper (big=true -> 128x128 tile, 8 waves)
    auto gemm = [&](int epi, bool transb, bool big,
                    const u16* A, long aO, long aI,
                    const u16* Bp, long bO, long bI,
                    const float* bias, long biasO, long biasI,
                    float* Cf, long cO, long cI,
                    u16* Cb, long cbO, long cbI,
                    int M, int N, int K, int lda, int ldb, int ldc,
                    int batch, int zmod) {
        const int bm = big ? 128 : 64;
        dim3 g((unsigned)(N / bm), (unsigned)(M / bm), (unsigned)batch), blk(bm * 2);
        if (big) {
            if (epi == 1)
                gemm_bf16<1, false, 128><<<g, blk, 0, stream>>>(A, aO, aI, Bp, bO, bI, bias, biasO, biasI,
                                                                Cf, cO, cI, Cb, cbO, cbI, M, N, K, lda, ldb, ldc, zmod);
            else
                gemm_bf16<0, false, 128><<<g, blk, 0, stream>>>(A, aO, aI, Bp, bO, bI, bias, biasO, biasI,
                                                                Cf, cO, cI, Cb, cbO, cbI, M, N, K, lda, ldb, ldc, zmod);
        } else {
            if (transb)
                gemm_bf16<0, true, 64><<<g, blk, 0, stream>>>(A, aO, aI, Bp, bO, bI, bias, biasO, biasI,
                                                              Cf, cO, cI, Cb, cbO, cbI, M, N, K, lda, ldb, ldc, zmod);
            else
                gemm_bf16<0, false, 64><<<g, blk, 0, stream>>>(A, aO, aI, Bp, bO, bI, bias, biasO, biasI,
                                                               Cf, cO, cI, Cb, cbO, cbI, M, N, K, lda, ldb, ldc, zmod);
        }
    };

    // --- pooling: logits[b] = selector [64x768] * latent[b]^T [768x4096]
    gemm(0, true, false, sel_b, 0, 0, lat_b, (long)Nc * Dc, 0, nullptr, 0, 0,
         logit_f, (long)Mc * Nc, 0, nullptr, 0, 0,
         Mc, Nc, Dc, Dc, Dc, Nc, Bc, 1);
    row_softmax<<<dim3(Bc * Mc), dim3(256), 0, stream>>>(logit_f, poolw_b);
    // x[b] = weights[b] [64x4096] * latent[b] [4096x768]
    gemm(0, false, false, poolw_b, (long)Mc * Nc, 0, lat_b, (long)Nc * Dc, 0, nullptr, 0, 0,
         x_f, (long)Mc * Dc, 0, x_b, (long)Mc * Dc, 0,
         Mc, Dc, Nc, Nc, Dc, Dc, Bc, 1);

    const long rows = (long)Bc * Mc;  // 512
    for (int l = 0; l < Lc; ++l) {
        // queries = qc[l] + x
        add_qc<<<dim3((unsigned)((rows * Dc + 255) / 256)), dim3(256), 0, stream>>>(
            x_f, qc + (long)l * Mc * Dc, qry_f, qry_b, (long)Mc * Dc, rows * Dc);
        // q = queries @ Wq + qb   (bf16 only)
        gemm(0, false, true, qry_b, 0, 0, wq_b + (long)l * Dc * Dc, 0, 0, qb + l * Dc, 0, 0,
             nullptr, 0, 0, q_b, 0, 0, (int)rows, Dc, Dc, Dc, Dc, Dc, 1, 1);
        // k = latent @ Wk + kb    (dominant GEMM, 32768x768x768)
        gemm(0, false, true, lat_b, 0, 0, wk_b + (long)l * Dc * Dc, 0, 0, kb + l * Dc, 0, 0,
             nullptr, 0, 0, k_b, 0, 0, Bc * Nc, Dc, Dc, Dc, Dc, Dc, 1, 1);
        // v = latent @ Wv + vb
        gemm(0, false, true, lat_b, 0, 0, wv_b + (long)l * Dc * Dc, 0, 0, vb + l * Dc, 0, 0,
             nullptr, 0, 0, v_b, 0, 0, Bc * Nc, Dc, Dc, Dc, Dc, Dc, 1, 1);
        // flash attention over 4096 keys
        flash_attn<<<dim3(Bc * Hc), dim3(128), 0, stream>>>(q_b, k_b, v_b, o_b, Hc, Nc, 0.125f);
        // head mix: per (b,h): o[64x64] @ hw[l,h][64x64] + hb -> [b,m,(h,f)]
        gemm(0, false, false, o_b, (long)Hc * 64 * 64, (long)64 * 64,
             whw_b + (long)l * Hc * 64 * 64, 0, (long)64 * 64,
             hb + (long)l * Hc * 64, 0, 64,
             om_f, (long)Mc * Dc, 64, om_b, (long)Mc * Dc, 64,
             Mc, 64, 64, 64, 64, Dc, Bc * Hc, Hc);
        // output projection
        gemm(0, false, true, om_b, 0, 0, wow_b + (long)l * Dc * Dc, 0, 0, obi + l * Dc, 0, 0,
             ao_f, 0, 0, nullptr, 0, 0, (int)rows, Dc, Dc, Dc, Dc, Dc, 1, 1);
        // x1 = LN(o + queries)
        ln_res<<<dim3((unsigned)rows), dim3(256), 0, stream>>>(
            ao_f, qry_f, g1 + l * Dc, be1 + l * Dc, x1_f, x1_b);
        // mlp hidden = gelu(x1 @ W1 + b1)  (bf16 only)
        gemm(1, false, true, x1_b, 0, 0, w1_b + (long)l * Dc * 4 * Dc, 0, 0, b1 + l * 4 * Dc, 0, 0,
             nullptr, 0, 0, mh_b, 0, 0, (int)rows, 4 * Dc, Dc, Dc, 4 * Dc, 4 * Dc, 1, 1);
        // mlp out = hidden @ W2 + b2
        gemm(0, false, true, mh_b, 0, 0, w2_b + (long)l * 4 * Dc * Dc, 0, 0, b2 + l * Dc, 0, 0,
             ml_f, 0, 0, nullptr, 0, 0, (int)rows, Dc, 4 * Dc, 4 * Dc, Dc, Dc, 1, 1);
        // x = LN(x1 + mlp); last layer writes d_out directly (fp32)
        float* xo = (l == Lc - 1) ? (float*)d_out : x_f;
        ln_res<<<dim3((unsigned)rows), dim3(256), 0, stream>>>(
            ml_f, x1_f, g2 + l * Dc, be2 + l * Dc, xo, x_b);
    }
}
